// CTC_70626442215693
// MI455X (gfx1250) — matrix operand order repaired and checked by reading, not compiled, not run on hardware
//
#include <hip/hip_runtime.h>
#include <hip/hip_bf16.h>
#include <stdint.h>

// ---------------------------------------------------------------------------
// CTC loss for MI455X (gfx1250, wave32, WMMA, async-to-LDS).
//
// Phase 0: one-time fp32 -> bf16 conversion of hs and W into workspace
//          (halves streamed bytes, removes convert-VALU from GEMM hot loops,
//          enables byte-exact async DMA staging).
// Phase 1: fused GEMM (hs @ W.T + b) + online logsumexp over V  -> lse[16000]
//          v_wmma_f32_16x16x32_bf16, f32 accumulate, double-buffered
//          GLOBAL_LOAD_ASYNC_TO_LDS staging of W K-chunks.
// Phase 2: gathered GEMM against W[ext[b,s],:]  -> lp[b,t,s] (201 states,
//          padded to 208), same async double-buffered staging.
// Phase 3: CTC alpha recursion (sequential in T) per batch.
// Phase 4: mean over batch.
// ---------------------------------------------------------------------------

#define NEGINF (-1e30f)

typedef __attribute__((ext_vector_type(16))) __bf16 v16bf;
typedef __attribute__((ext_vector_type(8)))  float  v8f;
typedef __attribute__((ext_vector_type(4)))  int    v4i;

typedef __attribute__((address_space(1))) v4i  glb_v4i;
typedef __attribute__((address_space(3))) v4i  lds_v4i;
typedef __attribute__((address_space(3))) char lds_char;

union Frag {
  uint4 q[2];
  v16bf v;
};

#if defined(__has_builtin)
#if __has_builtin(__builtin_amdgcn_global_load_async_to_lds_b128) && \
    __has_builtin(__builtin_amdgcn_s_wait_asynccnt)
#define USE_ASYNC_LDS 1
#endif
#endif
#ifndef USE_ASYNC_LDS
#define USE_ASYNC_LDS 0
#endif

// Copy 16 bytes global -> LDS (per lane). Async DMA when available.
__device__ __forceinline__ void copy16(const char* g, char* l) {
#if USE_ASYNC_LDS
  // AS3 dest: addrspacecast first (aperture-correct), then bitcast to v4i.
  __builtin_amdgcn_global_load_async_to_lds_b128(
      (glb_v4i*)(uintptr_t)g, (lds_v4i*)(lds_char*)l, 0, 0);
#else
  *(uint4*)l = *(const uint4*)g;
#endif
}
#if USE_ASYNC_LDS
#define ASYNC_WAIT(n) __builtin_amdgcn_s_wait_asynccnt(n)
#else
#define ASYNC_WAIT(n)
#endif

__device__ __forceinline__ unsigned short f2bf(float f) {
  unsigned u = __float_as_uint(f);
  u += 0x7FFFu + ((u >> 16) & 1u);  // round-to-nearest-even
  return (unsigned short)(u >> 16);
}
__device__ __forceinline__ unsigned pack2(float a, float b) {
  return (unsigned)f2bf(a) | ((unsigned)f2bf(b) << 16);
}

// ---------------------------------------------------------------------------
// Kernel 0: bulk fp32 -> bf16 (4 elems / thread).
// ---------------------------------------------------------------------------
__global__ void k_f32_to_bf16(const float* __restrict__ src,
                              unsigned short* __restrict__ dst, int n4)
{
  const int i = blockIdx.x * blockDim.x + threadIdx.x;
  if (i < n4) {
    const float4 v = ((const float4*)src)[i];
    uint2 p; p.x = pack2(v.x, v.y); p.y = pack2(v.z, v.w);
    ((uint2*)dst)[i] = p;
  }
}

// Stage one [64 n][32 k] bf16 W chunk into LDS (128 threads, 2 x 16B each).
// LDS layout: 64 contiguous bytes per column n (K-halves at +0 / +32),
// which is exactly a contiguous 64B slice of a bf16 W row -> pure byte copy.
__device__ __forceinline__ void stage_chunk64(const char* W16, char* dst,
                                              int nBase, int kc, int tid)
{
#pragma unroll
  for (int u = 0; u < 2; ++u) {
    const int task = tid + u * 128;        // 0..255
    const int n = task >> 2, q = task & 3; // 16B quarter of the 64B row slice
    copy16(W16 + (size_t)(nBase + n) * 1024 + kc * 2 + q * 16,
           dst + n * 64 + q * 16);
  }
}

// ---------------------------------------------------------------------------
// Kernel 1: logits GEMM + fused online logsumexp over V.
//   M = 16000 (b*t), K = 512 (E), N = 4000 (V).
//   250 WGs x 128 threads (4 waves). 64-row M tile resident in LDS (bf16),
//   N swept in 64-col tiles, K inner in 32-chunks, W chunks double-buffered
//   via async-to-LDS so DMA overlaps the WMMA stream.
// ---------------------------------------------------------------------------
__global__ void k_gemm_lse(const char* __restrict__ hs16,
                           const char* __restrict__ W16,
                           const float* __restrict__ bias,
                           float* __restrict__ lse)
{
  extern __shared__ char smem[];
  char* ldsA = smem;               // [64 rows][512 k] bf16 = 65536 B
  char* ldsB = smem + 64 * 1024;   // 2 x 4096 B double buffer

  const int  tid   = threadIdx.x;
  const int  lane  = tid & 31;
  const int  wave  = tid >> 5;
  const bool lo    = lane < 16;
  const int  mBase = blockIdx.x * 64;

  // Stage A tile: 4096 x 16B tasks (byte copy, bf16 already).
  for (int u = 0; u < 32; ++u) {
    const int task = tid + u * 128;
    const int row = task >> 6, q = task & 63;
    copy16(hs16 + (size_t)(mBase + row) * 1024 + q * 16,
           ldsA + row * 1024 + q * 16);
  }
  ASYNC_WAIT(0);
  __syncthreads();

  float mrun[8], srun[8];
#pragma unroll
  for (int i = 0; i < 8; ++i) { mrun[i] = NEGINF; srun[i] = 0.f; }

  const int rowA = wave * 16 + (lane & 15);

  int buf = 0;
  stage_chunk64(W16, ldsB, 0, 0, tid);  // prologue: first chunk into buf 0

  for (int nBase = 0; nBase < 4000; nBase += 64) {
    v8f c[4];
#pragma unroll
    for (int j = 0; j < 4; ++j)
#pragma unroll
      for (int i = 0; i < 8; ++i) c[j][i] = 0.f;

    for (int kc = 0; kc < 512; kc += 32) {
      // Issue next chunk into the other buffer, then wait for current.
      int nn = nBase, nk = kc + 32;
      if (nk == 512) { nn += 64; nk = 0; }
      if (nn < 4000) {
        stage_chunk64(W16, ldsB + (buf ^ 1) * 4096, nn, nk, tid);
        ASYNC_WAIT(2);   // current chunk (2 older asyncs/wave) complete
      } else {
        ASYNC_WAIT(0);
      }
      __syncthreads();

      Frag a;
      {
        const char* base = ldsA + rowA * 1024 + kc * 2 + (lo ? 0 : 16);
        a.q[0] = *(const uint4*)(base);
        a.q[1] = *(const uint4*)(base + 32);
      }
      const char* bcur = ldsB + buf * 4096;
#pragma unroll
      for (int j = 0; j < 4; ++j) {
        Frag bf;
        const char* bb = bcur + (j * 16 + (lane & 15)) * 64 + (lo ? 0 : 16);
        bf.q[0] = *(const uint4*)(bb);
        bf.q[1] = *(const uint4*)(bb + 32);
        c[j] = __builtin_amdgcn_wmma_f32_16x16x32_bf16(
            false, a.v, false, bf.v, (short)0, c[j], false, false);
      }
      buf ^= 1;
      __syncthreads();  // all frag reads done before next stage overwrites
    }

    // Bias + online logsumexp (register/shuffle only; no LDS).
#pragma unroll
    for (int j = 0; j < 4; ++j) {
      const float bj = bias[nBase + j * 16 + (lane & 15)];
#pragma unroll
      for (int i = 0; i < 8; ++i) c[j][i] += bj;
    }
#pragma unroll
    for (int i = 0; i < 8; ++i) {
      float tmax = fmaxf(fmaxf(c[0][i], c[1][i]), fmaxf(c[2][i], c[3][i]));
#pragma unroll
      for (int m = 1; m < 16; m <<= 1) tmax = fmaxf(tmax, __shfl_xor(tmax, m, 32));
      const float newm = fmaxf(mrun[i], tmax);
      float part = __expf(c[0][i] - newm) + __expf(c[1][i] - newm) +
                   __expf(c[2][i] - newm) + __expf(c[3][i] - newm);
#pragma unroll
      for (int m = 1; m < 16; m <<= 1) part += __shfl_xor(part, m, 32);
      srun[i] = srun[i] * __expf(mrun[i] - newm) + part;
      mrun[i] = newm;
    }
  }

  if ((lane & 15) == 0) {
#pragma unroll
    for (int i = 0; i < 8; ++i) {
      const int row = wave * 16 + (lo ? i : i + 8);
      lse[mBase + row] = mrun[i] + __logf(srun[i]);
    }
  }
}

// ---------------------------------------------------------------------------
// Kernel 2: gathered logits at extended-label states.
//   lp[b,t,s] = dot(hs[b,t,:], W[ext[b,s],:]) + bias[ext] - lse[b*T+t]
//   Grid (8 t-tiles, 16 batches) x 256 threads (8 waves x 16 rows = 128 t).
//   N = 256 staged columns (208 used), 13 WMMA subtiles, double-buffered.
// ---------------------------------------------------------------------------
__global__ void k_gemm_labels(const char* __restrict__ hs16,
                              const char* __restrict__ W16,
                              const float* __restrict__ bias,
                              const int* __restrict__ ys,
                              const float* __restrict__ lse,
                              float* __restrict__ lp)
{
  extern __shared__ char smem[];
  char* ldsA = smem;                            // 128 x 1024 B = 131072 B
  char* ldsB = smem + 128 * 1024;               // 2 x 16384 B double buffer
  int*  ldsE = (int*)(smem + 128 * 1024 + 32768);  // ext indices [256]

  const int  tid   = threadIdx.x;
  const int  lane  = tid & 31;
  const int  wave  = tid >> 5;
  const bool lo    = lane < 16;
  const int  b     = blockIdx.y;
  const int  tBase = blockIdx.x * 128;

  {
    int v = 0;
    if (tid < 201 && (tid & 1)) v = ys[b * 100 + (tid >> 1)];
    ldsE[tid] = v;  // 201..255 -> blank row 0 (padding, never read back)
  }

  // Stage A tile (rows clamped to t=999 for 1000->1024 padding).
  for (int u = 0; u < 32; ++u) {
    const int task = tid + u * 256;   // 0..8191
    const int row = task >> 6, q = task & 63;
    int gt = tBase + row; if (gt > 999) gt = 999;
    copy16(hs16 + ((size_t)b * 1000 + gt) * 1024 + q * 16,
           ldsA + row * 1024 + q * 16);
  }
  ASYNC_WAIT(0);
  __syncthreads();   // A visible, ldsE visible

  v8f c[13];
#pragma unroll
  for (int j = 0; j < 13; ++j)
#pragma unroll
    for (int i = 0; i < 8; ++i) c[j][i] = 0.f;

  const int rowA = wave * 16 + (lane & 15);

  // Gathered B chunk stage: 1024 x 16B tasks (4 / thread, uniform per wave).
  auto stage_gather = [&](char* dst, int kc) {
#pragma unroll
    for (int u = 0; u < 4; ++u) {
      const int task = tid + u * 256;        // 0..1023
      const int n = task >> 2, q = task & 3; // n in 0..255
      copy16(W16 + (size_t)ldsE[n] * 1024 + kc * 2 + q * 16,
             dst + n * 64 + q * 16);
    }
  };

  int buf = 0;
  stage_gather(ldsB, 0);

  for (int kc = 0; kc < 512; kc += 32) {
    if (kc + 32 < 512) {
      stage_gather(ldsB + (buf ^ 1) * 16384, kc + 32);
      ASYNC_WAIT(4);
    } else {
      ASYNC_WAIT(0);
    }
    __syncthreads();

    Frag a;
    {
      const char* base = ldsA + rowA * 1024 + kc * 2 + (lo ? 0 : 16);
      a.q[0] = *(const uint4*)(base);
      a.q[1] = *(const uint4*)(base + 32);
    }
    const char* bcur = ldsB + buf * 16384;
#pragma unroll
    for (int j = 0; j < 13; ++j) {
      Frag bf;
      const char* bb = bcur + (j * 16 + (lane & 15)) * 64 + (lo ? 0 : 16);
      bf.q[0] = *(const uint4*)(bb);
      bf.q[1] = *(const uint4*)(bb + 32);
      c[j] = __builtin_amdgcn_wmma_f32_16x16x32_bf16(
          false, a.v, false, bf.v, (short)0, c[j], false, false);
    }
    buf ^= 1;
    __syncthreads();
  }

  // Epilogue: + bias[ext] - lse, store lp[b][t][s] (s-contiguous).
  float lsev[8];
#pragma unroll
  for (int i = 0; i < 8; ++i) {
    const int gt = tBase + wave * 16 + (lo ? i : i + 8);
    lsev[i] = (gt < 1000) ? lse[b * 1000 + gt] : 0.f;
  }
#pragma unroll
  for (int j = 0; j < 13; ++j) {
    const int s  = j * 16 + (lane & 15);
    const float bj = bias[ldsE[s]];
#pragma unroll
    for (int i = 0; i < 8; ++i) {
      const int gt = tBase + wave * 16 + (lo ? i : i + 8);
      const float out = (gt < 1000) ? (c[j][i] + bj - lsev[i]) : NEGINF;
      lp[(((size_t)b << 10) + gt) * 208 + s] = out;
    }
  }
}

// ---------------------------------------------------------------------------
// Kernel 3: CTC alpha recursion. One workgroup per batch, thread s = state.
// ---------------------------------------------------------------------------
__global__ void k_ctc_dp(const float* __restrict__ lp,
                         const int* __restrict__ ys,
                         float* __restrict__ nll)
{
  __shared__ float alpha[208];
  const int  b   = blockIdx.x;
  const int  s   = threadIdx.x;
  const bool act = s < 201;

  int  extv = 0;
  bool skip = false;
  if (act) {
    extv = (s & 1) ? ys[b * 100 + (s >> 1)] : 0;
    if (s >= 2) {
      const int ext2 = ((s - 2) & 1) ? ys[b * 100 + ((s - 2) >> 1)] : 0;
      skip = (extv != 0) && (extv != ext2);
    }
  }

  const float* lpb = lp + ((size_t)b << 10) * 208;
  if (s < 208) alpha[s] = (act && s < 2) ? lpb[s] : NEGINF;
  __syncthreads();

  for (int t = 1; t < 1000; ++t) {
    float nv = NEGINF;
    if (act) {
      const float a1 = alpha[s];
      const float a2 = (s >= 1) ? alpha[s - 1] : NEGINF;
      const float a3 = skip ? alpha[s - 2] : NEGINF;
      const float m  = fmaxf(fmaxf(a1, a2), a3);
      const float sum = __expf(a1 - m) + __expf(a2 - m) + __expf(a3 - m);
      nv = m + __logf(sum) + lpb[(size_t)t * 208 + s];
    }
    __syncthreads();
    if (act) alpha[s] = nv;
    __syncthreads();
  }

  if (s == 0) {
    const float x = alpha[200], y = alpha[199];
    const float m = fmaxf(x, y);
    nll[b] = -(m + __logf(__expf(x - m) + __expf(y - m)));
  }
}

// ---------------------------------------------------------------------------
// Kernel 4: mean over batch.
// ---------------------------------------------------------------------------
__global__ void k_finalize(const float* __restrict__ nll, float* __restrict__ out)
{
  if (threadIdx.x == 0 && blockIdx.x == 0) {
    float s = 0.f;
    for (int i = 0; i < 16; ++i) s += nll[i];
    out[0] = s * (1.0f / 16.0f);
  }
}

// ---------------------------------------------------------------------------
extern "C" void kernel_launch(void* const* d_in, const int* in_sizes, int n_in,
                              void* d_out, int out_size, void* d_ws, size_t ws_size,
                              hipStream_t stream)
{
  const float* hs   = (const float*)d_in[0];  // [16,1000,512]
  const int*   ys   = (const int*)  d_in[1];  // [16,100]
  const float* W    = (const float*)d_in[2];  // [4000,512]
  const float* bias = (const float*)d_in[3];  // [4000]
  float*       out  = (float*)d_out;

  char*  ws   = (char*)d_ws;
  float* lse  = (float*)ws;                                  // 16000 f32
  float* nll  = (float*)(ws + 64 * 1024);                    // 16 f32
  char*  hs16 = ws + 128 * 1024;                             // 16,384,000 B
  char*  W16  = hs16 + (size_t)16000 * 512 * 2;              // 4,096,000 B
  float* lp   = (float*)(W16 + (size_t)4000 * 512 * 2);      // 13,631,488 B

  // Phase 0: one-time bf16 materialization.
  k_f32_to_bf16<<<dim3(8000), dim3(256), 0, stream>>>(
      hs, (unsigned short*)hs16, 16000 * 512 / 4);
  k_f32_to_bf16<<<dim3(2000), dim3(256), 0, stream>>>(
      W, (unsigned short*)W16, 4000 * 512 / 4);

  const size_t lds1 = 64 * 1024 + 2 * 4096;                  // 73,728 B
  k_gemm_lse<<<dim3(250), dim3(128), lds1, stream>>>(hs16, W16, bias, lse);

  const size_t lds2 = 128 * 1024 + 2 * 16384 + 256 * 4;      // 164,864 B
  k_gemm_labels<<<dim3(8, 16), dim3(256), lds2, stream>>>(hs16, W16, bias, ys,
                                                          lse, lp);

  k_ctc_dp<<<dim3(16), dim3(256), 0, stream>>>(lp, ys, nll);
  k_finalize<<<dim3(1), dim3(64), 0, stream>>>(nll, out);
}
